// RPN_13082470383789
// MI455X (gfx1250) — matrix operand order repaired and checked by reading, not compiled, not run on hardware
//
#include <hip/hip_runtime.h>
#include <math.h>

typedef __attribute__((ext_vector_type(16))) _Float16 v16h;
typedef __attribute__((ext_vector_type(8)))  float    v8f;
typedef __attribute__((ext_vector_type(4)))  unsigned u32x4;
typedef __attribute__((ext_vector_type(8)))  int      i32x8;
typedef __attribute__((ext_vector_type(4)))  int      i32x4;

#define HH 64
#define WW 64
#define CIN 2048
#define COUT 512
#define NP (HH*WW)          // 4096 spatial positions
#define NANCH 9
#define NPROB (NANCH*NP)    // 36864
#define TOPK 2000
#define NMSW 63             // ceil(2016/32) u32 words per row
#define KC 64               // K halves per LDS tile (2 WMMA K-steps)
#define PITCH 72            // halves per LDS row (64 data + 8 pad, keeps 16B align)
#define NIT (9 * (CIN / KC))// 288 pipeline steps (9 taps x 32 K-chunks)

#define A_HALVES (64 * PITCH)    // 4608 halves
#define B_HALVES (128 * PITCH)   // 9216 halves per buffer

// Tensor Data Mover availability (probe-verified builtin names; arity differs
// between ROCm 7.2 clang-22 (5 args) and clang-23 toolchain (6 args)).
#if defined(__has_builtin)
#  if __has_builtin(__builtin_amdgcn_tensor_load_to_lds) && \
      __has_builtin(__builtin_amdgcn_s_wait_tensorcnt)
#    define USE_TDM 1
#  endif
#endif
#ifndef USE_TDM
#  define USE_TDM 0
#endif

// ---------- layout prep: x (NCHW f32) -> xh (HWC f16) ----------
__global__ void k_cvt_x(const float* __restrict__ x, _Float16* __restrict__ xh) {
  int idx = blockIdx.x * 256 + threadIdx.x;          // idx = p*CIN + c
  if (idx >= CIN * NP) return;
  int c = idx & (CIN - 1);
  int p = idx >> 11;
  xh[idx] = (_Float16)x[c * NP + p];
}

// ---------- layout prep: conv_w (OIHW f32) -> wt[tap][o][c] f16 ----------
__global__ void k_cvt_w(const float* __restrict__ w, _Float16* __restrict__ wt) {
  int idx = blockIdx.x * 256 + threadIdx.x;          // idx = (tap*COUT+o)*CIN + c
  if (idx >= 9 * COUT * CIN) return;
  int c   = idx & (CIN - 1);
  int o   = (idx >> 11) & (COUT - 1);
  int tap = idx / (CIN * COUT);
  wt[idx] = (_Float16)w[o * (CIN * 9) + c * 9 + tap];
}

#if USE_TDM
// Issue a TDM load of one B tile: 2D tile 64 halves x 128 rows from wt into LDS,
// with pad_interval=128B / pad_amount=4 dwords reproducing the 72-half pitch.
__device__ __forceinline__ void tdm_load_b(const _Float16* gsrc, unsigned lds_byte_off) {
  unsigned long long ga = (unsigned long long)(const void*)gsrc;
  u32x4 g0;
  g0[0] = 1u;                                            // count=1 (valid user D#)
  g0[1] = lds_byte_off;                                  // lds_addr [63:32]
  g0[2] = (unsigned)(ga & 0xFFFFFFFFu);                  // global_addr [95:64]
  g0[3] = (unsigned)((ga >> 32) & 0x01FFFFFFu) | (2u << 30); // addr[56:32] | type=2
  i32x8 g1;
  g1[0] = (int)((1u << 16)      // data_size = 2 bytes
              | (1u << 20)      // pad_enable
              | (4u << 22)      // pad_interval: pad after 128 bytes (one row)
              | (3u << 25));    // pad_amount: 4 dwords (8 halves) -> pitch 72
  g1[1] = (int)(((unsigned)CIN & 0xFFFFu) << 16);        // tensor_dim0[15:0] @ [63:48]
  g1[2] = (int)((((unsigned)CIN >> 16) & 0xFFFFu)        // tensor_dim0[31:16]
              | ((9u * COUT & 0xFFFFu) << 16));          // tensor_dim1[15:0]
  g1[3] = (int)(((9u * COUT >> 16) & 0xFFFFu)            // tensor_dim1[31:16]
              | (64u << 16));                            // tile_dim0 = 64 halves
  g1[4] = (int)128u;                                     // tile_dim1 = 128 rows
  g1[5] = (int)CIN;                                      // dim0_stride[31:0]
  g1[6] = (int)(((unsigned)CIN & 0xFFFFu) << 16);        // dim0_stride[47:32]=0 | dim1_stride[15:0]
  g1[7] = (int)(((unsigned)CIN >> 16) & 0xFFFFu);        // dim1_stride[47:16]
  i32x4 z4 = {0, 0, 0, 0};
#if __clang_major__ >= 23
  i32x8 z8 = {0, 0, 0, 0, 0, 0, 0, 0};
  __builtin_amdgcn_tensor_load_to_lds(g0, g1, z4, z4, z8, 0);
#else
  __builtin_amdgcn_tensor_load_to_lds(g0, g1, z4, z4, 0);
#endif
}
#endif

// ---------- 3x3 conv as 9 shifted GEMMs, WMMA f32<=f16xf16, bias+ReLU ----------
// Block = 256 thr (8 waves). Block tile: M=64 (x positions of row y), N=128 out-ch.
// Wave grid 2(M) x 4(N); each wave owns a 32x32 D tile = 4 accumulators.
// A tiles: register-pipelined global loads (halo zero-fill). B tiles: TDM
// double-buffered async DMA (TENSORcnt) when available, else same reg pipeline.
__global__ __launch_bounds__(256) void k_conv_wmma(
    const _Float16* __restrict__ xh, const _Float16* __restrict__ wt,
    const float* __restrict__ bias, float* __restrict__ feat) {
#if USE_TDM
  __shared__ _Float16 SH[A_HALVES + 2 * B_HALVES];   // A + double-buffered B (46 KB)
  _Float16* Ash = SH;
#else
  __shared__ _Float16 Ash[A_HALVES];
  __shared__ _Float16 Bsh[B_HALVES];
#endif

  const int tid  = threadIdx.x;
  const int lane = tid & 31;
  const int wave = tid >> 5;
  const int wm   = wave & 1;              // M half (32 rows)
  const int wn   = wave >> 1;             // N quarter (32 cols)
  const int y    = blockIdx.x;            // spatial row
  const int n0   = blockIdx.y * 128;      // out-channel base

  const int frowA = tid >> 2;             // A fill: 4 thr/row, 16 halves each
  const int fqA   = tid & 3;

  v8f acc00 = {0.f,0.f,0.f,0.f,0.f,0.f,0.f,0.f};
  v8f acc01 = acc00, acc10 = acc00, acc11 = acc00;

  u32x4 aR[2];
  auto gloadA = [&](int it) {
    const int tap = it >> 5;              // 32 K-chunks per tap
    const int kc  = (it & 31) << 6;       // K base within the 2048 channels
    const int yy  = y + tap / 3 - 1;
    const int xx  = frowA + tap % 3 - 1;
    const bool inb = (yy >= 0) && (yy < HH) && (xx >= 0) && (xx < WW);
    const u32x4 z = {0u, 0u, 0u, 0u};
    aR[0] = z; aR[1] = z;
    if (inb) {
      const _Float16* as = xh + (size_t)(yy * WW + xx) * CIN + kc + fqA * 16;
      aR[0] = *(const u32x4*)(as);
      aR[1] = *(const u32x4*)(as + 8);
      __builtin_prefetch(as + KC, 0, 0);              // gfx1250 global_prefetch_b8
    }
  };

#if !USE_TDM
  const int frowB = tid >> 1;             // B fill: 2 thr/row, 32 halves each
  const int fqB   = tid & 1;
  u32x4 bR[4];
  auto gloadB = [&](int it) {
    const int tap = it >> 5;
    const int kc  = (it & 31) << 6;
    const _Float16* bs = wt + (size_t)(tap * COUT + n0 + frowB) * CIN + kc + fqB * 32;
    bR[0] = *(const u32x4*)(bs);
    bR[1] = *(const u32x4*)(bs + 8);
    bR[2] = *(const u32x4*)(bs + 16);
    bR[3] = *(const u32x4*)(bs + 24);
    __builtin_prefetch(bs + KC, 0, 0);
  };
#endif

  gloadA(0);
#if USE_TDM
  if (wave == 0) {
    // Pre-issue tile 0 into B buffer 0 via the Tensor Data Mover.
    tdm_load_b(wt + (size_t)(0 * COUT + n0) * CIN + 0, (unsigned)(A_HALVES * 2));
  }
#else
  gloadB(0);
#endif

  #pragma unroll 1
  for (int it = 0; it < NIT; ++it) {
    __syncthreads();                       // previous tile fully consumed
    *(u32x4*)(&Ash[frowA * PITCH + fqA * 16])     = aR[0];
    *(u32x4*)(&Ash[frowA * PITCH + fqA * 16 + 8]) = aR[1];
#if USE_TDM
    if (wave == 0) {
      if (it + 1 < NIT) {
        const int tap1 = (it + 1) >> 5;
        const int kc1  = ((it + 1) & 31) << 6;
        const unsigned dstoff = (unsigned)((A_HALVES + ((it + 1) & 1) * B_HALVES) * 2);
        tdm_load_b(wt + (size_t)(tap1 * COUT + n0) * CIN + kc1, dstoff);
        __builtin_amdgcn_s_wait_tensorcnt(1);  // in-order: current tile resident
      } else {
        __builtin_amdgcn_s_wait_tensorcnt(0);
      }
    }
    const _Float16* Bcur = SH + A_HALVES + (it & 1) * B_HALVES;
#else
    *(u32x4*)(&Bsh[frowB * PITCH + fqB * 32])      = bR[0];
    *(u32x4*)(&Bsh[frowB * PITCH + fqB * 32 + 8])  = bR[1];
    *(u32x4*)(&Bsh[frowB * PITCH + fqB * 32 + 16]) = bR[2];
    *(u32x4*)(&Bsh[frowB * PITCH + fqB * 32 + 24]) = bR[3];
    const _Float16* Bcur = Bsh;
#endif
    __syncthreads();                       // LDS tiles visible to all waves

    if (it + 1 < NIT) {                    // load-ahead overlaps the 8 WMMAs below
      gloadA(it + 1);
#if !USE_TDM
      gloadB(it + 1);
#endif
    }

    // Fragments per CDNA5 16-bit A(16x32)/B(32x16) VGPR layouts (CDNA5 ISA):
    // lane<16 gets K 0..7 & 16..23, lane>=16 gets K 8..15 & 24..31 (A);
    // B: lane n=lane%16, same K split as A. All 16B ds_load_b128.
    const int hi = lane >> 4;
    const int lr = lane & 15;
    #pragma unroll
    for (int ko2 = 0; ko2 < 2; ++ko2) {
      const int ko = ko2 * 32;
      union Frag { u32x4 u[2]; v16h h; } a0, a1, b0, b1;
      const int am0 = wm * 32 + lr;
      a0.u[0] = *(const u32x4*)(&Ash[am0 * PITCH + ko + hi * 8]);
      a0.u[1] = *(const u32x4*)(&Ash[am0 * PITCH + ko + hi * 8 + 16]);
      const int am1 = am0 + 16;
      a1.u[0] = *(const u32x4*)(&Ash[am1 * PITCH + ko + hi * 8]);
      a1.u[1] = *(const u32x4*)(&Ash[am1 * PITCH + ko + hi * 8 + 16]);
      const int bn0 = wn * 32 + lr;
      b0.u[0] = *(const u32x4*)(&Bcur[bn0 * PITCH + ko + hi * 8]);
      b0.u[1] = *(const u32x4*)(&Bcur[bn0 * PITCH + ko + hi * 8 + 16]);
      const int bn1 = bn0 + 16;
      b1.u[0] = *(const u32x4*)(&Bcur[bn1 * PITCH + ko + hi * 8]);
      b1.u[1] = *(const u32x4*)(&Bcur[bn1 * PITCH + ko + hi * 8 + 16]);

      acc00 = __builtin_amdgcn_wmma_f32_16x16x32_f16(false, a0.h, false, b0.h,
                                                     (short)0, acc00, false, false);
      acc01 = __builtin_amdgcn_wmma_f32_16x16x32_f16(false, a0.h, false, b1.h,
                                                     (short)0, acc01, false, false);
      acc10 = __builtin_amdgcn_wmma_f32_16x16x32_f16(false, a1.h, false, b0.h,
                                                     (short)0, acc10, false, false);
      acc11 = __builtin_amdgcn_wmma_f32_16x16x32_f16(false, a1.h, false, b1.h,
                                                     (short)0, acc11, false, false);
    }
  }

  // D layout: VGPR v, lane l -> m = v + (l>=16 ? 8:0), n = l%16. Bias + ReLU.
  const int hi = lane >> 4;
  const int lr = lane & 15;
  const int cn0 = n0 + wn * 32 + lr;
  const float bia0 = bias[cn0];
  const float bia1 = bias[cn0 + 16];
  #pragma unroll
  for (int v = 0; v < 8; ++v) {
    const int m0 = wm * 32 + v + hi * 8;        // spatial x, sub-tile 0
    const int m1 = m0 + 16;                     // sub-tile 1
    const int p0 = y * WW + m0;
    const int p1 = y * WW + m1;
    float t00 = acc00[v] + bia0; t00 = t00 > 0.f ? t00 : 0.f;
    float t01 = acc01[v] + bia1; t01 = t01 > 0.f ? t01 : 0.f;
    float t10 = acc10[v] + bia0; t10 = t10 > 0.f ? t10 : 0.f;
    float t11 = acc11[v] + bia1; t11 = t11 > 0.f ? t11 : 0.f;
    feat[(size_t)p0 * COUT + cn0]      = t00;
    feat[(size_t)p0 * COUT + cn0 + 16] = t01;
    feat[(size_t)p1 * COUT + cn0]      = t10;
    feat[(size_t)p1 * COUT + cn0 + 16] = t11;
  }
}

// ---------- 1x1 heads: 54 dot products of length 512 per position ----------
__global__ void k_heads(const float* __restrict__ feat,
                        const float* __restrict__ cls_w, const float* __restrict__ cls_b,
                        const float* __restrict__ bbox_w, const float* __restrict__ bbox_b,
                        float* __restrict__ headout) {
  int idx = blockIdx.x * 256 + threadIdx.x;
  if (idx >= 54 * NP) return;
  int p = idx & (NP - 1);
  int j = idx >> 12;
  const float* wrow = (j < 18) ? (cls_w + j * COUT) : (bbox_w + (j - 18) * COUT);
  float b = (j < 18) ? cls_b[j] : bbox_b[j - 18];
  const float* f = feat + (size_t)p * COUT;
  float acc = 0.f;
  #pragma unroll 4
  for (int c = 0; c < COUT; ++c) acc += f[c] * wrow[c];
  headout[j * NP + p] = acc + b;
}

// ---------- 2-class softmax -> foreground prob (sigmoid of logit diff) ----------
__global__ void k_probs(const float* __restrict__ headout, float* __restrict__ probs) {
  int idx = blockIdx.x * 256 + threadIdx.x;
  if (idx >= NPROB) return;
  int a = idx >> 12, p = idx & (NP - 1);
  float l0 = headout[a * NP + p];
  float l1 = headout[(9 + a) * NP + p];
  probs[idx] = 1.f / (1.f + expf(l0 - l1));
}

// ---------- exact sorted top-2000 via global rank (tie-break: lower index) ----------
__global__ __launch_bounds__(256) void k_topk(const float* __restrict__ probs,
                                              float* __restrict__ ts, int* __restrict__ ti) {
  __shared__ float cache[256];
  int i = blockIdx.x * 256 + threadIdx.x;
  float s = probs[i];
  int rank = 0;
  for (int base = 0; base < NPROB; base += 256) {
    __syncthreads();
    cache[threadIdx.x] = probs[base + threadIdx.x];
    __syncthreads();
    #pragma unroll 8
    for (int t = 0; t < 256; ++t) {
      float v = cache[t];
      int   j = base + t;
      rank += (v > s || (v == s && j < i)) ? 1 : 0;
    }
  }
  if (rank < TOPK) { ts[rank] = s; ti[rank] = i; }
}

// ---------- anchor gen + delta decode for the 2000 kept proposals ----------
__global__ void k_decode(const int* __restrict__ ti, const float* __restrict__ headout,
                         float* __restrict__ boxes) {
  int i = blockIdx.x * 256 + threadIdx.x;
  if (i >= TOPK) return;
  int idx = ti[i];
  int a = idx >> 12, p = idx & (NP - 1);
  int si = a / 3, ri = a % 3;
  float scale = (float)(8 << si);            // 8,16,32
  float ratio = 0.5f * (float)(1 << ri);     // 0.5,1,2
  float sr = sqrtf(ratio);
  float aw = scale * sr;
  float ah = scale / sr;
  float cx = (float)((p & 63) * 32);
  float cy = (float)((p >> 6) * 32);
  float dx = headout[(18 + a * 4 + 0) * NP + p];
  float dy = headout[(18 + a * 4 + 1) * NP + p];
  float dw = headout[(18 + a * 4 + 2) * NP + p];
  float dh = headout[(18 + a * 4 + 3) * NP + p];
  float pcx = cx + dx * aw;
  float pcy = cy + dy * ah;
  float pw = expf(dw) * aw;
  float ph = expf(dh) * ah;
  boxes[i * 4 + 0] = pcx - 0.5f * pw;
  boxes[i * 4 + 1] = pcy - 0.5f * ph;
  boxes[i * 4 + 2] = pcx + 0.5f * pw;
  boxes[i * 4 + 3] = pcy + 0.5f * ph;
}

__global__ void k_zero(unsigned* __restrict__ p, int n) {
  int i = blockIdx.x * 256 + threadIdx.x;
  if (i < n) p[i] = 0u;
}

// ---------- suppression bitmask: row i, bit j set if iou(i,j)>thr and j>i ----------
__global__ void k_sup(const float* __restrict__ boxes, unsigned* __restrict__ sup) {
  int i = blockIdx.x;
  float ix1 = boxes[i * 4 + 0], iy1 = boxes[i * 4 + 1];
  float ix2 = boxes[i * 4 + 2], iy2 = boxes[i * 4 + 3];
  float ai = (ix2 - ix1) * (iy2 - iy1);
  for (int j = threadIdx.x; j < TOPK; j += 256) {
    if (j <= i) continue;
    float jx1 = boxes[j * 4 + 0], jy1 = boxes[j * 4 + 1];
    float jx2 = boxes[j * 4 + 2], jy2 = boxes[j * 4 + 3];
    float aj = (jx2 - jx1) * (jy2 - jy1);
    float w = fmaxf(fminf(ix2, jx2) - fmaxf(ix1, jx1), 0.f);
    float h = fmaxf(fminf(iy2, jy2) - fmaxf(iy1, jy1), 0.f);
    float inter = w * h;
    float iou = inter / (ai + aj - inter + 1e-9f);
    if (iou > 0.5f) atomicOr(&sup[i * NMSW + (j >> 5)], 1u << (j & 31));
  }
}

// ---------- serial NMS sweep (matches fori_loop semantics) + masked output ----------
__global__ __launch_bounds__(64) void k_nms_out(const unsigned* __restrict__ sup,
                                                const float* __restrict__ boxes,
                                                const float* __restrict__ scores,
                                                float* __restrict__ out) {
  __shared__ unsigned keep[NMSW];
  int t = threadIdx.x;
  if (t < NMSW) keep[t] = (t == NMSW - 1) ? 0x0000FFFFu : 0xFFFFFFFFu; // only j<2000 valid
  __syncthreads();
  for (int i = 0; i < TOPK; ++i) {
    bool alive = (keep[i >> 5] >> (i & 31)) & 1u;
    __syncthreads();
    if (alive && t < NMSW) keep[t] &= ~sup[i * NMSW + t];
    __syncthreads();
  }
  for (int i = t; i < TOPK; i += 64) {
    float k = ((keep[i >> 5] >> (i & 31)) & 1u) ? 1.f : 0.f;
    out[i * 4 + 0] = boxes[i * 4 + 0] * k;
    out[i * 4 + 1] = boxes[i * 4 + 1] * k;
    out[i * 4 + 2] = boxes[i * 4 + 2] * k;
    out[i * 4 + 3] = boxes[i * 4 + 3] * k;
    out[4 * TOPK + i] = scores[i] * k;
  }
}

extern "C" void kernel_launch(void* const* d_in, const int* in_sizes, int n_in,
                              void* d_out, int out_size, void* d_ws, size_t ws_size,
                              hipStream_t stream) {
  const float* x      = (const float*)d_in[0];
  const float* conv_w = (const float*)d_in[1];
  const float* conv_b = (const float*)d_in[2];
  const float* cls_w  = (const float*)d_in[3];
  const float* cls_b  = (const float*)d_in[4];
  const float* bbox_w = (const float*)d_in[5];
  const float* bbox_b = (const float*)d_in[6];
  float* out = (float*)d_out;

  char* ws = (char*)d_ws;
  size_t off = 0;
  auto carve = [&](size_t bytes) -> void* {
    void* p = ws + off;
    off = (off + bytes + 255) & ~(size_t)255;
    return p;
  };
  _Float16* xh      = (_Float16*)carve((size_t)CIN * NP * 2);       // 16 MB
  _Float16* wt      = (_Float16*)carve((size_t)9 * COUT * CIN * 2); // 18 MB
  float*    feat    = (float*)carve((size_t)NP * COUT * 4);         // 8 MB
  float*    headout = (float*)carve((size_t)54 * NP * 4);
  float*    probs   = (float*)carve((size_t)NPROB * 4);
  float*    ts      = (float*)carve((size_t)TOPK * 4);
  int*      ti      = (int*)carve((size_t)TOPK * 4);
  float*    boxes   = (float*)carve((size_t)TOPK * 16);
  unsigned* sup     = (unsigned*)carve((size_t)TOPK * NMSW * 4);

  k_cvt_x<<<(CIN * NP + 255) / 256, 256, 0, stream>>>(x, xh);
  k_cvt_w<<<(9 * COUT * CIN + 255) / 256, 256, 0, stream>>>(conv_w, wt);
  k_conv_wmma<<<dim3(HH, COUT / 128), 256, 0, stream>>>(xh, wt, conv_b, feat);
  k_heads<<<(54 * NP + 255) / 256, 256, 0, stream>>>(feat, cls_w, cls_b, bbox_w, bbox_b, headout);
  k_probs<<<NPROB / 256, 256, 0, stream>>>(headout, probs);
  k_topk<<<NPROB / 256, 256, 0, stream>>>(probs, ts, ti);
  k_decode<<<(TOPK + 255) / 256, 256, 0, stream>>>(ti, headout, boxes);
  k_zero<<<(TOPK * NMSW + 255) / 256, 256, 0, stream>>>(sup, TOPK * NMSW);
  k_sup<<<TOPK, 256, 0, stream>>>(boxes, sup);
  k_nms_out<<<1, 64, 0, stream>>>(sup, boxes, ts, out);
}
